// MHAttention_15358803050646
// MI455X (gfx1250) — matrix operand order repaired and checked by reading, not compiled, not run on hardware
//
#include <hip/hip_runtime.h>
#include <hip/hip_bf16.h>

// ---------------------------------------------------------------------------
// CDNA5 / gfx1250 MHA (buggy-einsum variant)
// wave32, v_wmma_f32_16x16x32_bf16, double-buffered LDS, global prefetch,
// packed fp32->bf16 conversion when the hardware builtin is available.
// ---------------------------------------------------------------------------

typedef __attribute__((ext_vector_type(16))) __bf16 v16bf;
typedef __attribute__((ext_vector_type(8)))  __bf16 v8bf;
typedef __attribute__((ext_vector_type(2)))  __bf16 v2bf;
typedef __attribute__((ext_vector_type(8)))  float  v8f;

union Frag16 { v16bf v; v8bf h[2]; };

__device__ __forceinline__ unsigned short f2bf(float f) {
    union { float f; unsigned u; } x; x.f = f;
    unsigned r = x.u + 0x7FFFu + ((x.u >> 16) & 1u);   // round-to-nearest-even
    return (unsigned short)(r >> 16);
}

// two fp32 -> packed bf16x2 in one dword (hardware v_cvt_pk_bf16_f32 if present)
__device__ __forceinline__ unsigned pk_bf16(float x, float y) {
#if __has_builtin(__builtin_amdgcn_cvt_pk_bf16_f32)
    union { v2bf v; unsigned u; } r;
    r.v = __builtin_amdgcn_cvt_pk_bf16_f32(x, y);
    return r.u;
#else
    return (unsigned)f2bf(x) | ((unsigned)f2bf(y) << 16);
#endif
}

__device__ __forceinline__ v8f wmma_bf16(v16bf a, v16bf b, v8f c) {
    // D = A(16x32 bf16) x B(32x16 bf16) + C(16x16 f32)
    return __builtin_amdgcn_wmma_f32_16x16x32_bf16(
        /*neg_a=*/false, a, /*neg_b=*/false, b,
        /*c_mod=*/(short)0, c, /*reuse_a=*/false, /*reuse_b=*/false);
}

// ---------------------------------------------------------------------------
// C[b] = A(MxK) * B[b](KxN) + bias. Block 256 threads (8 waves), tile 128x256,
// K-step 32, double-buffered LDS. Waves 2(M) x 4(N), each 64x64 = 4x4 WMMA
// accumulators -> 16 ds_load_b128 : 16 wmma per k-step.
// Requires M%128==0, N%256==0, K%32==0.
// ---------------------------------------------------------------------------
#define GBM 128
#define GBN 256
#define GKB 32

__global__ __launch_bounds__(256)
void gemm_bias_bf16(const float* __restrict__ A, const float* __restrict__ B,
                    const float* __restrict__ bias, float* __restrict__ C,
                    int M, int N, int K) {
    __shared__ __attribute__((aligned(16))) unsigned short sA[2][GBM * GKB]; // [row][k]
    __shared__ __attribute__((aligned(16))) unsigned short sB[2][GBN * GKB]; // [n][k] (B^T)

    const int tid = threadIdx.x;
    const int bn  = blockIdx.x * GBN;
    const int bm  = blockIdx.y * GBM;
    const float* Bb = B + (size_t)blockIdx.z * K * N;
    float*       Cb = C + (size_t)blockIdx.z * M * N;

    const int wid = tid >> 5, lane = tid & 31;
    const int lh = lane >> 4, ln = lane & 15;
    const int wm = (wid >> 2) * 64;   // 2 waves over 128 rows
    const int wn = (wid & 3) * 64;    // 4 waves over 256 cols

    float4 ra[4], rb[8];              // in-flight staging registers

    auto stage = [&](int k0) {        // issue global loads for tile at k0
        #pragma unroll
        for (int j = 0; j < 4; ++j) {
            int i = tid + j * 256;
            int r = i >> 3, kg = (i & 7) << 2;
            ra[j] = *(const float4*)(A + (size_t)(bm + r) * K + k0 + kg);
        }
        #pragma unroll
        for (int j = 0; j < 8; ++j) {
            int i = tid + j * 256;
            int kk = i >> 6, ng = (i & 63) << 2;
            rb[j] = *(const float4*)(Bb + (size_t)(k0 + kk) * N + bn + ng);
        }
    };
    auto commit = [&](int buf) {      // convert + store staged tile into LDS
        #pragma unroll
        for (int j = 0; j < 4; ++j) {
            int i = tid + j * 256;
            int r = i >> 3, kg = (i & 7) << 2;
            uint2 pk; pk.x = pk_bf16(ra[j].x, ra[j].y);
                      pk.y = pk_bf16(ra[j].z, ra[j].w);
            *(uint2*)&sA[buf][r * GKB + kg] = pk;    // 8-byte ds_store
        }
        #pragma unroll
        for (int j = 0; j < 8; ++j) {                // transposed scatter (n-major)
            int i = tid + j * 256;
            int kk = i >> 6, ng = (i & 63) << 2;
            unsigned p01 = pk_bf16(rb[j].x, rb[j].y);
            unsigned p23 = pk_bf16(rb[j].z, rb[j].w);
            sB[buf][(ng + 0) * GKB + kk] = (unsigned short)p01;
            sB[buf][(ng + 1) * GKB + kk] = (unsigned short)(p01 >> 16);
            sB[buf][(ng + 2) * GKB + kk] = (unsigned short)p23;
            sB[buf][(ng + 3) * GKB + kk] = (unsigned short)(p23 >> 16);
        }
    };

    v8f acc[4][4] = {};

    stage(0);
    commit(0);
    __syncthreads();
    int cur = 0;

    for (int k0 = 0; k0 < K; k0 += GKB) {
        const bool hasNext = (k0 + GKB) < K;
        if (hasNext) {
            stage(k0 + GKB);                         // overlap HBM/L2 with WMMA
            if (k0 + 2 * GKB < K) {                  // hint tile after next
                __builtin_prefetch(A + (size_t)(bm + (tid >> 3)) * K + k0 + 2 * GKB, 0, 1);
                __builtin_prefetch(Bb + (size_t)(k0 + 2 * GKB + (tid >> 6)) * N
                                        + bn + ((tid & 63) << 2), 0, 1);
            }
        }

        // ---- compute on buffer `cur` ----
        v16bf bfrag[4];
        #pragma unroll
        for (int nt = 0; nt < 4; ++nt) {   // lane col n; elems e -> K = lh*16 + e
            const unsigned short* p = &sB[cur][(wn + nt * 16 + ln) * GKB + lh * 16];
            Frag16 f; f.h[0] = *(const v8bf*)p; f.h[1] = *(const v8bf*)(p + 8);
            bfrag[nt] = f.v;
        }
        #pragma unroll
        for (int mt = 0; mt < 4; ++mt) {   // lane row m; elems e -> K = lh*16 + e
            const unsigned short* p = &sA[cur][(wm + mt * 16 + ln) * GKB];
            Frag16 f;
            f.h[0] = *(const v8bf*)(p + lh * 16);
            f.h[1] = *(const v8bf*)(p + lh * 16 + 8);
            #pragma unroll
            for (int nt = 0; nt < 4; ++nt)
                acc[mt][nt] = wmma_bf16(f.v, bfrag[nt], acc[mt][nt]);
        }

        if (hasNext) {
            commit(cur ^ 1);
            __syncthreads();
            cur ^= 1;
        }
    }

    // epilogue: lane ln = column, VGPR v -> row v + 8*lh
    #pragma unroll
    for (int mt = 0; mt < 4; ++mt) {
        #pragma unroll
        for (int nt = 0; nt < 4; ++nt) {
            const int col = bn + wn + nt * 16 + ln;
            #pragma unroll
            for (int v = 0; v < 8; ++v) {
                const int row = bm + wm + mt * 16 + v + 8 * lh;
                Cb[(size_t)row * N + col] = acc[mt][nt][v] + bias[row];
            }
        }
    }
}

// ---------------------------------------------------------------------------
// Attention, faithful to 'bnqk,bnqd->bnqd': out[q,d] = V[q,d] * sum_k softmax(S)[q,k]
// Flash-style stats-only pass: S = (Q/8) K^T via WMMA (scale folded into Q
// staging), online row max/sum over double-buffered 128-key tiles.
// grid.x = 8 (q blocks of 128), grid.y = 64 (b*8 + head). 8 waves, 16 q rows each.
// ---------------------------------------------------------------------------
__global__ __launch_bounds__(256)
void attn_rowsum_bf16(const float* __restrict__ qkv, float* __restrict__ out) {
    __shared__ __attribute__((aligned(16))) unsigned short sQ[128 * 64];
    __shared__ __attribute__((aligned(16))) unsigned short sK[2][128 * 64];

    const int bh = blockIdx.y;
    const int b  = bh >> 3;
    const int n  = bh & 7;
    const size_t base = (size_t)b * 1536 * 1024 + (size_t)n * 65536;
    const float* Q  = qkv + base;                  // (1024 x 64) row-major
    const float* Kp = qkv + base + 512 * 1024;
    const float* V  = qkv + base + 1024 * 1024;
    float* O = out + (size_t)b * 512 * 1024 + (size_t)n * 65536;

    const int q0  = blockIdx.x * 128;
    const int tid = threadIdx.x;
    const int wid = tid >> 5, lane = tid & 31;
    const int lh = lane >> 4, ln = lane & 15;

    float4 rk[8];
    auto stageK = [&](int kt) {
        #pragma unroll
        for (int j = 0; j < 8; ++j) {
            int i = tid + j * 256;
            int r = i >> 4, dg = (i & 15) << 2;
            rk[j] = *(const float4*)(Kp + (size_t)(kt + r) * 64 + dg);
        }
    };
    auto commitK = [&](int buf) {
        #pragma unroll
        for (int j = 0; j < 8; ++j) {
            int i = tid + j * 256;
            int r = i >> 4, dg = (i & 15) << 2;
            uint2 pk; pk.x = pk_bf16(rk[j].x, rk[j].y);
                      pk.y = pk_bf16(rk[j].z, rk[j].w);
            *(uint2*)&sK[buf][r * 64 + dg] = pk;
        }
    };

    // stage Q block (128 x 64) with 1/sqrt(d_k) folded in
    for (int i = tid; i < 2048; i += 256) {
        int r = i >> 4, dg = (i & 15) << 2;
        float4 f = *(const float4*)(Q + (size_t)(q0 + r) * 64 + dg);
        uint2 pk; pk.x = pk_bf16(f.x * 0.125f, f.y * 0.125f);
                  pk.y = pk_bf16(f.z * 0.125f, f.w * 0.125f);
        *(uint2*)&sQ[r * 64 + dg] = pk;
    }
    stageK(0);
    commitK(0);
    __syncthreads();
    int cur = 0;

    float run_m[8], run_s[8];
    #pragma unroll
    for (int v = 0; v < 8; ++v) { run_m[v] = -__builtin_inff(); run_s[v] = 0.0f; }

    for (int kt = 0; kt < 1024; kt += 128) {
        const bool hasNext = (kt + 128) < 1024;
        if (hasNext) stageK(kt + 128);

        // S tile: 16 q rows (this wave) x 128 keys, K-dim 64 -> 2 steps x 8 tiles
        v8f acc[8] = {};
        #pragma unroll
        for (int kk = 0; kk < 64; kk += 32) {
            const unsigned short* pa = &sQ[(wid * 16 + ln) * 64 + kk];
            Frag16 fa;
            fa.h[0] = *(const v8bf*)(pa + lh * 16);
            fa.h[1] = *(const v8bf*)(pa + lh * 16 + 8);
            #pragma unroll
            for (int nt = 0; nt < 8; ++nt) {
                const unsigned short* pb = &sK[cur][(nt * 16 + ln) * 64 + kk + lh * 16];
                Frag16 fb;
                fb.h[0] = *(const v8bf*)(pb);
                fb.h[1] = *(const v8bf*)(pb + 8);
                acc[nt] = wmma_bf16(fa.v, fb.v, acc[nt]);
            }
        }

        // online softmax stats per row (row = v + 8*lh of this wave's 16)
        #pragma unroll
        for (int v = 0; v < 8; ++v) {
            float m = -3.0e38f;
            #pragma unroll
            for (int nt = 0; nt < 8; ++nt) m = fmaxf(m, acc[nt][v]);
            #pragma unroll
            for (int off = 1; off < 16; off <<= 1) m = fmaxf(m, __shfl_xor(m, off));
            const float newm = fmaxf(run_m[v], m);
            float s = 0.0f;
            #pragma unroll
            for (int nt = 0; nt < 8; ++nt) s += __expf(acc[nt][v] - newm);
            #pragma unroll
            for (int off = 1; off < 16; off <<= 1) s += __shfl_xor(s, off);
            run_s[v] = run_s[v] * __expf(run_m[v] - newm) + s;
            run_m[v] = newm;
        }

        if (hasNext) {
            commitK(cur ^ 1);
            __syncthreads();
            cur ^= 1;
        }
    }

    // out[q,d] = V[q,d] * (D / D)  — softmax row-sum, numerically ~1
    #pragma unroll
    for (int v = 0; v < 8; ++v) {
        const float D = run_s[v];
        const float f = D / D;
        const int q = q0 + wid * 16 + v + 8 * lh;
        for (int d = ln; d < 64; d += 16)
            O[(size_t)q * 64 + d] = V[(size_t)q * 64 + d] * f;
    }
}

// ---------------------------------------------------------------------------
extern "C" void kernel_launch(void* const* d_in, const int* in_sizes, int n_in,
                              void* d_out, int out_size, void* d_ws, size_t ws_size,
                              hipStream_t stream) {
    (void)in_sizes; (void)n_in; (void)out_size; (void)ws_size;
    const float* x     = (const float*)d_in[0];   // (8, 512, 32, 32)
    const float* qkv_w = (const float*)d_in[1];   // (1536, 512)
    const float* qkv_b = (const float*)d_in[2];   // (1536)
    const float* out_w = (const float*)d_in[3];   // (512, 512)
    const float* out_b = (const float*)d_in[4];   // (512)
    float* out = (float*)d_out;                   // (8, 512, 32, 32)

    float* qkv = (float*)d_ws;                    // (8, 1536, 1024) fp32
    float* att = qkv + (size_t)8 * 1536 * 1024;   // (8, 512, 1024) fp32

    // 1) qkv = qkv_w @ x + qkv_b   (per batch: 1536x512 * 512x1024)
    dim3 g1(1024 / GBN, 1536 / GBM, 8);
    gemm_bias_bf16<<<g1, 256, 0, stream>>>(qkv_w, x, qkv_b, qkv, 1536, 1024, 512);

    // 2) attention row-sum pass (buggy einsum => scaled V passthrough)
    dim3 g2(8, 64);
    attn_rowsum_bf16<<<g2, 256, 0, stream>>>(qkv, att);

    // 3) out = out_w @ att + out_b (per batch: 512x512 * 512x1024)
    dim3 g3(1024 / GBN, 512 / GBM, 8);
    gemm_bias_bf16<<<g3, 256, 0, stream>>>(out_w, att, out_b, out, 512, 1024, 512);
}
